// LongExtTransformerEncoder_31877247271676
// MI455X (gfx1250) — compile-verified, matrix operand order checked by reading, not run on hardware
//
#include <hip/hip_runtime.h>
#include <hip/hip_bf16.h>
#include <math.h>

// ---------------------------------------------------------------------------
// Problem constants (reference: B,S,H,NH,D,L,W,G,F = 2,4096,768,12,64,12,256,64,3072)
// ---------------------------------------------------------------------------
constexpr int Bz = 2;
constexpr int Sq = 4096;
constexpr int Hd = 768;
constexpr int NHh = 12;
constexpr int Dd = 64;
constexpr int Ll = 12;
constexpr int Wn = 256;     // local window
constexpr int Gg = 64;      // global tokens
constexpr int Ff = 3072;
constexpr int nCh = Sq / Wn; // 16 chunks
constexpr float NEGV = -1.0e9f;

typedef __attribute__((ext_vector_type(16))) _Float16 v16h;
typedef __attribute__((ext_vector_type(8)))  _Float16 v8h;
typedef __attribute__((ext_vector_type(8)))  float    v8f;

// ---------------------------------------------------------------------------
// WMMA helpers (CDNA5 16x16x32 f16 -> f32)
// ---------------------------------------------------------------------------
__device__ __forceinline__ v8f wmma_f16(v16h a, v16h b, v8f c) {
  return __builtin_amdgcn_wmma_f32_16x16x32_f16(false, a, false, b, (short)0, c,
                                                false, false);
}

// Load a 16x32 fp16 fragment (A layout; also valid for B when the tile is
// stored n-major / k-contiguous).  Per CDNA5 ISA 7.12.2:
//   lanes 0-15  : row = lane,    halves = K 0..7  then K 16..23
//   lanes 16-31 : row = lane-16, halves = K 8..15 then K 24..31
__device__ __forceinline__ v16h load_frag16x32(const _Float16* base, int ld) {
  int lane = threadIdx.x & 31;
  int m = lane & 15;
  int kg = (lane >> 4) & 1;
  const _Float16* p = base + (size_t)m * ld + kg * 8;
  v8h lo = *(const v8h*)(p);
  v8h hi = *(const v8h*)(p + 16);
  return __builtin_shufflevector(lo, hi, 0, 1, 2, 3, 4, 5, 6, 7, 8, 9, 10, 11,
                                 12, 13, 14, 15);
}

// ---------------------------------------------------------------------------
// Elementwise / setup kernels
// ---------------------------------------------------------------------------
__global__ void am_kernel(const float* __restrict__ ext, float* __restrict__ am,
                          int n) {
  int i = blockIdx.x * 256 + threadIdx.x;
  if (i < n) am[i] = (ext[i] < 0.0f) ? NEGV : 0.0f;
}

__global__ __launch_bounds__(256) void embed_kernel(
    const float* __restrict__ sv, const unsigned char* __restrict__ mask,
    float* __restrict__ x, _Float16* __restrict__ xh) {
  int row = blockIdx.x;            // b*S + s
  int s = row & (Sq - 1);
  float mk = mask[row] ? 1.0f : 0.0f;
  const float sqrtH = 27.712812921102035f;       // sqrt(768)
  for (int c = threadIdx.x; c < Hd; c += 256) {
    int j = c >> 1;
    float div = __expf((float)(2 * j) * (-9.210340371976184f / (float)Hd));
    float ang = (float)s * div;
    float pe = (c & 1) ? __cosf(ang) : __sinf(ang);
    float v = sv[(size_t)row * Hd + c] * mk * sqrtH + pe;
    x[(size_t)row * Hd + c] = v;
    xh[(size_t)row * Hd + c] = (_Float16)v;
  }
}

// LayerNorm over H=768; one row per block, 256 threads, 3 elems/thread.
__global__ __launch_bounds__(256) void ln_kernel(
    const float* __restrict__ x, const float* __restrict__ g,
    const float* __restrict__ bta, float* __restrict__ outF,
    _Float16* __restrict__ outH) {
  __shared__ float red[8];
  __shared__ float stat;
  int row = blockIdx.x, tid = threadIdx.x;
  const float* xp = x + (size_t)row * Hd;
  float v0 = xp[tid], v1 = xp[tid + 256], v2 = xp[tid + 512];
  float s = v0 + v1 + v2;
#pragma unroll
  for (int off = 16; off >= 1; off >>= 1) s += __shfl_xor(s, off, 32);
  if ((tid & 31) == 0) red[tid >> 5] = s;
  __syncthreads();
  if (tid == 0) {
    float t = 0.0f;
    for (int i = 0; i < 8; ++i) t += red[i];
    stat = t * (1.0f / (float)Hd);
  }
  __syncthreads();
  float mu = stat;
  float d0 = v0 - mu, d1 = v1 - mu, d2 = v2 - mu;
  float q = d0 * d0 + d1 * d1 + d2 * d2;
#pragma unroll
  for (int off = 16; off >= 1; off >>= 1) q += __shfl_xor(q, off, 32);
  __syncthreads();
  if ((tid & 31) == 0) red[tid >> 5] = q;
  __syncthreads();
  if (tid == 0) {
    float t = 0.0f;
    for (int i = 0; i < 8; ++i) t += red[i];
    stat = rsqrtf(t * (1.0f / (float)Hd) + 1e-6f);
  }
  __syncthreads();
  float rs = stat;
  float y0 = d0 * rs * g[tid] + bta[tid];
  float y1 = d1 * rs * g[tid + 256] + bta[tid + 256];
  float y2 = d2 * rs * g[tid + 512] + bta[tid + 512];
  if (outF) {
    outF[(size_t)row * Hd + tid] = y0;
    outF[(size_t)row * Hd + tid + 256] = y1;
    outF[(size_t)row * Hd + tid + 512] = y2;
  }
  if (outH) {
    outH[(size_t)row * Hd + tid] = (_Float16)y0;
    outH[(size_t)row * Hd + tid + 256] = (_Float16)y1;
    outH[(size_t)row * Hd + tid + 512] = (_Float16)y2;
  }
}

// ---------------------------------------------------------------------------
// Generic WMMA GEMM: out = epilogue(A(f16, MxK) * W(f32, KxN) + bias)
// Block: 256 threads / 8 waves, 64x64 tile; wave -> 16(M) x 32(N).
// MODE: 0 = bias, 1 = (bias)*scale, 2 = bias+GELU, 3 = bias+residual
// ---------------------------------------------------------------------------
template <int MODE>
__global__ __launch_bounds__(256) void gemm_kernel(
    const _Float16* __restrict__ A, const float* __restrict__ Wt,
    const float* __restrict__ bias, float* outF, _Float16* outH,
    const float* res, int M, int N, int K, float scale) {
  __shared__ _Float16 ldsB[64 * 40];  // [n][k] k-contiguous, pad 40
  int tid = threadIdx.x;
  int lane = tid & 31, w = tid >> 5;
  int wm = w & 3, wn = w >> 2;
  int m0 = blockIdx.y * 64, n0 = blockIdx.x * 64;
  v8f acc0 = {}, acc1 = {};
  const _Float16* arow = A + (size_t)(m0 + wm * 16) * K;
  for (int k0 = 0; k0 < K; k0 += 32) {
    __syncthreads();
    // stage 32(K) x 64(N) of W into LDS transposed, converting f32 -> f16
    int e0 = tid * 8;
    int kk = e0 >> 6;       // 0..31
    int nn = e0 & 63;       // multiple of 8
    const float* wp = Wt + (size_t)(k0 + kk) * N + n0 + nn;
#pragma unroll
    for (int i = 0; i < 8; ++i) ldsB[(nn + i) * 40 + kk] = (_Float16)wp[i];
    __syncthreads();
    __builtin_prefetch(arow + k0 + 64, 0, 1);   // global_prefetch_b8
    v16h a = load_frag16x32(arow + k0, K);
    v16h b0 = load_frag16x32(ldsB + (wn * 32 + 0) * 40, 40);
    v16h b1 = load_frag16x32(ldsB + (wn * 32 + 16) * 40, 40);
    acc0 = wmma_f16(a, b0, acc0);
    acc1 = wmma_f16(a, b1, acc1);
  }
  int hi = lane >> 4, ln = lane & 15;
#pragma unroll
  for (int nt = 0; nt < 2; ++nt) {
    v8f acc = nt ? acc1 : acc0;
    int col = n0 + wn * 32 + nt * 16 + ln;
    float bv = bias[col];
#pragma unroll
    for (int r = 0; r < 8; ++r) {
      int row = m0 + wm * 16 + r + 8 * hi;
      float v = acc[r] + bv;
      if (MODE == 1) v *= scale;
      if (MODE == 2) v = 0.5f * v * (1.0f + erff(v * 0.7071067811865475f));
      if (MODE == 3) v += res[(size_t)row * N + col];
      if (outF) outF[(size_t)row * N + col] = v;
      if (outH) outH[(size_t)row * N + col] = (_Float16)v;
    }
  }
}

// ---------------------------------------------------------------------------
// Local (windowed) attention, flash-style.  Grid: (nC, NH, B), block 256.
// Wave w handles 32 queries (2 M-tiles); 26 key-tiles of 32:
//   tiles 0..1  -> global keys 0..63 (always valid, bias am[key])
//   tiles 2..25 -> local cols j = (tile-2)*32.., key = c*W - W + j,
//                  valid iff j in [qr, qr+2W] and key in [G, S)
// ---------------------------------------------------------------------------
__global__ __launch_bounds__(256) void local_attn_kernel(
    const _Float16* __restrict__ q, const _Float16* __restrict__ k,
    const _Float16* __restrict__ v, const float* __restrict__ am,
    _Float16* __restrict__ out) {
  __shared__ _Float16 ldsK[32 * 72];   // [key][d]  (B-frag for QK^T)
  __shared__ _Float16 ldsVT[64 * 40];  // [d][key]  (B-frag for PV)
  __shared__ _Float16 ldsP[8][32 * 40];
  __shared__ float ldsAm[32];
  int c = blockIdx.x, h = blockIdx.y, b = blockIdx.z;
  int tid = threadIdx.x, lane = tid & 31, w = tid >> 5;
  int qbase = w * 32;
  int hi = lane >> 4, ln = lane & 15;

  size_t baseQ = ((size_t)(b * Sq) + c * Wn + qbase) * Hd + h * Dd;
  v16h qf[2][2];
#pragma unroll
  for (int mt = 0; mt < 2; ++mt)
#pragma unroll
    for (int t = 0; t < 2; ++t)
      qf[mt][t] = load_frag16x32(q + baseQ + (size_t)mt * 16 * Hd + t * 32, Hd);

  v8f acc[2][4] = {};
  float mrow[2][8], lrow[2][8];
#pragma unroll
  for (int mt = 0; mt < 2; ++mt)
#pragma unroll
    for (int r = 0; r < 8; ++r) { mrow[mt][r] = -1e30f; lrow[mt][r] = 0.0f; }

  for (int tile = 0; tile < 26; ++tile) {
    bool isglob = tile < 2;
    int keybase = isglob ? tile * 32 : (c * Wn - Wn + (tile - 2) * 32);
    __syncthreads();
    {  // stage K (row-major) and V (transposed) + am
      int e0 = tid * 8;
      int kr = e0 >> 6;
      int d0 = e0 & 63;
      int key = keybase + kr;
      bool okk = (key >= 0) && (key < Sq);
      size_t gofs = ((size_t)(b * Sq) + (okk ? key : 0)) * Hd + h * Dd + d0;
      const _Float16* kp = k + gofs;
      const _Float16* vp = v + gofs;
#pragma unroll
      for (int i = 0; i < 8; ++i) {
        _Float16 kv = okk ? kp[i] : (_Float16)0.0f;
        _Float16 vv = okk ? vp[i] : (_Float16)0.0f;
        ldsK[kr * 72 + d0 + i] = kv;
        ldsVT[(d0 + i) * 40 + kr] = vv;
      }
      if (tid < 32) {
        int key2 = keybase + tid;
        ldsAm[tid] = (key2 >= 0 && key2 < Sq) ? am[b * Sq + key2] : NEGV;
      }
    }
    __syncthreads();

    v16h bK[2][2], bV[4];
#pragma unroll
    for (int nt = 0; nt < 2; ++nt)
#pragma unroll
      for (int t = 0; t < 2; ++t)
        bK[nt][t] = load_frag16x32(ldsK + nt * 16 * 72 + t * 32, 72);
#pragma unroll
    for (int dn = 0; dn < 4; ++dn)
      bV[dn] = load_frag16x32(ldsVT + dn * 16 * 40, 40);

#pragma unroll
    for (int mt = 0; mt < 2; ++mt) {
      v8f sc[2];
#pragma unroll
      for (int nt = 0; nt < 2; ++nt) {
        v8f z = {};
        z = wmma_f16(qf[mt][0], bK[nt][0], z);
        z = wmma_f16(qf[mt][1], bK[nt][1], z);
        sc[nt] = z;
      }
      // mask + bias
#pragma unroll
      for (int nt = 0; nt < 2; ++nt) {
        int colt = nt * 16 + ln;
        float amv = ldsAm[colt];
#pragma unroll
        for (int r = 0; r < 8; ++r) {
          int qr = qbase + mt * 16 + r + 8 * hi;
          float sv = sc[nt][r];
          if (isglob) {
            sv += amv;
          } else {
            int j = (tile - 2) * 32 + colt;
            int key = c * Wn - Wn + j;
            bool ok = (j >= qr) && (j <= qr + 2 * Wn) && (key >= Gg) && (key < Sq);
            sv = ok ? (sv + amv) : NEGV;
          }
          sc[nt][r] = sv;
        }
      }
      // online softmax per row
#pragma unroll
      for (int r = 0; r < 8; ++r) {
        float mx = fmaxf(sc[0][r], sc[1][r]);
#pragma unroll
        for (int off = 8; off >= 1; off >>= 1)
          mx = fmaxf(mx, __shfl_xor(mx, off, 32));
        float mnew = fmaxf(mrow[mt][r], mx);
        float f = __expf(mrow[mt][r] - mnew);
        mrow[mt][r] = mnew;
        float p0 = __expf(sc[0][r] - mnew);
        float p1 = __expf(sc[1][r] - mnew);
        float ps = p0 + p1;
#pragma unroll
        for (int off = 8; off >= 1; off >>= 1) ps += __shfl_xor(ps, off, 32);
        lrow[mt][r] = lrow[mt][r] * f + ps;
#pragma unroll
        for (int dn = 0; dn < 4; ++dn) acc[mt][dn][r] *= f;
        int prow = mt * 16 + r + 8 * hi;
        ldsP[w][prow * 40 + ln] = (_Float16)p0;
        ldsP[w][prow * 40 + 16 + ln] = (_Float16)p1;
      }
      v16h pf = load_frag16x32(&ldsP[w][mt * 16 * 40], 40);
#pragma unroll
      for (int dn = 0; dn < 4; ++dn)
        acc[mt][dn] = wmma_f16(pf, bV[dn], acc[mt][dn]);
    }
  }
  // write O
#pragma unroll
  for (int mt = 0; mt < 2; ++mt)
#pragma unroll
    for (int dn = 0; dn < 4; ++dn)
#pragma unroll
      for (int r = 0; r < 8; ++r) {
        int row = c * Wn + qbase + mt * 16 + r + 8 * hi;
        float ov = acc[mt][dn][r] / lrow[mt][r];
        out[((size_t)(b * Sq) + row) * Hd + h * Dd + dn * 16 + ln] =
            (_Float16)ov;
      }
}

// ---------------------------------------------------------------------------
// Global attention (rows s < G, dense over all S keys).  Grid (NH, B),
// block 128 (4 waves), wave w -> queries w*16..w*16+15.
// ---------------------------------------------------------------------------
__global__ __launch_bounds__(128) void global_attn_kernel(
    const _Float16* __restrict__ qg, const _Float16* __restrict__ kg,
    const _Float16* __restrict__ vg, const float* __restrict__ am,
    _Float16* __restrict__ out) {
  __shared__ _Float16 ldsK[32 * 72];
  __shared__ _Float16 ldsVT[64 * 40];
  __shared__ _Float16 ldsP[4][16 * 40];
  __shared__ float ldsAm[32];
  int h = blockIdx.x, b = blockIdx.y;
  int tid = threadIdx.x, lane = tid & 31, w = tid >> 5;
  int hi = lane >> 4, ln = lane & 15;

  size_t baseQ = ((size_t)(b * Sq) + w * 16) * Hd + h * Dd;
  v16h qf[2];
#pragma unroll
  for (int t = 0; t < 2; ++t)
    qf[t] = load_frag16x32(qg + baseQ + t * 32, Hd);

  v8f acc[4] = {};
  float mrow[8], lrow[8];
#pragma unroll
  for (int r = 0; r < 8; ++r) { mrow[r] = -1e30f; lrow[r] = 0.0f; }

  for (int kt = 0; kt < Sq / 32; ++kt) {
    int keybase = kt * 32;
    __syncthreads();
    {
      int e0 = tid * 16;
      int kr = e0 >> 6;
      int d0 = e0 & 63;
      int key = keybase + kr;
      size_t gofs = ((size_t)(b * Sq) + key) * Hd + h * Dd + d0;
      const _Float16* kp = kg + gofs;
      const _Float16* vp = vg + gofs;
#pragma unroll
      for (int i = 0; i < 16; ++i) {
        ldsK[kr * 72 + d0 + i] = kp[i];
        ldsVT[(d0 + i) * 40 + kr] = vp[i];
      }
      if (tid < 32) ldsAm[tid] = am[b * Sq + keybase + tid];
    }
    __syncthreads();

    v16h bK[2][2], bV[4];
#pragma unroll
    for (int nt = 0; nt < 2; ++nt)
#pragma unroll
      for (int t = 0; t < 2; ++t)
        bK[nt][t] = load_frag16x32(ldsK + nt * 16 * 72 + t * 32, 72);
#pragma unroll
    for (int dn = 0; dn < 4; ++dn)
      bV[dn] = load_frag16x32(ldsVT + dn * 16 * 40, 40);

    v8f sc[2];
#pragma unroll
    for (int nt = 0; nt < 2; ++nt) {
      v8f z = {};
      z = wmma_f16(qf[0], bK[nt][0], z);
      z = wmma_f16(qf[1], bK[nt][1], z);
      sc[nt] = z;
    }
#pragma unroll
    for (int nt = 0; nt < 2; ++nt) {
      float amv = ldsAm[nt * 16 + ln];
#pragma unroll
      for (int r = 0; r < 8; ++r) sc[nt][r] += amv;
    }
#pragma unroll
    for (int r = 0; r < 8; ++r) {
      float mx = fmaxf(sc[0][r], sc[1][r]);
#pragma unroll
      for (int off = 8; off >= 1; off >>= 1)
        mx = fmaxf(mx, __shfl_xor(mx, off, 32));
      float mnew = fmaxf(mrow[r], mx);
      float f = __expf(mrow[r] - mnew);
      mrow[r] = mnew;
      float p0 = __expf(sc[0][r] - mnew);
      float p1 = __expf(sc[1][r] - mnew);
      float ps = p0 + p1;
#pragma unroll
      for (int off = 8; off >= 1; off >>= 1) ps += __shfl_xor(ps, off, 32);
      lrow[r] = lrow[r] * f + ps;
#pragma unroll
      for (int dn = 0; dn < 4; ++dn) acc[dn][r] *= f;
      int prow = r + 8 * hi;
      ldsP[w][prow * 40 + ln] = (_Float16)p0;
      ldsP[w][prow * 40 + 16 + ln] = (_Float16)p1;
    }
    v16h pf = load_frag16x32(&ldsP[w][0], 40);
#pragma unroll
    for (int dn = 0; dn < 4; ++dn) acc[dn] = wmma_f16(pf, bV[dn], acc[dn]);
  }
#pragma unroll
  for (int dn = 0; dn < 4; ++dn)
#pragma unroll
    for (int r = 0; r < 8; ++r) {
      int srow = w * 16 + r + 8 * hi;  // < 64 = G
      float ov = acc[dn][r] / lrow[r];
      out[((size_t)(b * Sq) + srow) * Hd + h * Dd + dn * 16 + ln] =
          (_Float16)ov;
    }
}

// ---------------------------------------------------------------------------
// Host driver
// ---------------------------------------------------------------------------
extern "C" void kernel_launch(void* const* d_in, const int* in_sizes, int n_in,
                              void* d_out, int out_size, void* d_ws,
                              size_t ws_size, hipStream_t stream) {
  (void)in_sizes; (void)n_in; (void)out_size; (void)ws_size;
  // setup_inputs order: sent_vecs, mask, extended_mask, params.
  // params tree-flattened with sorted dict keys:
  //  3 final_b, 4 final_g, then layers (alphabetical):
  //  5 W1, 6 W2, 7 Wk, 8 Wkg, 9 Wo, 10 Wq, 11 Wqg, 12 Wv, 13 Wvg,
  //  14 b1, 15 b2, 16 bk, 17 bkg, 18 bo, 19 bq, 20 bqg, 21 bv, 22 bvg,
  //  23 ln1_b, 24 ln1_g, 25 ln2_b, 26 ln2_g
  const float* sent = (const float*)d_in[0];
  const unsigned char* mask = (const unsigned char*)d_in[1];
  const float* extm = (const float*)d_in[2];
  const float* final_b = (const float*)d_in[3];
  const float* final_g = (const float*)d_in[4];
  const float* pW1 = (const float*)d_in[5];
  const float* pW2 = (const float*)d_in[6];
  const float* pWk = (const float*)d_in[7];
  const float* pWkg = (const float*)d_in[8];
  const float* pWo = (const float*)d_in[9];
  const float* pWq = (const float*)d_in[10];
  const float* pWqg = (const float*)d_in[11];
  const float* pWv = (const float*)d_in[12];
  const float* pWvg = (const float*)d_in[13];
  const float* pb1 = (const float*)d_in[14];
  const float* pb2 = (const float*)d_in[15];
  const float* pbk = (const float*)d_in[16];
  const float* pbkg = (const float*)d_in[17];
  const float* pbo = (const float*)d_in[18];
  const float* pbq = (const float*)d_in[19];
  const float* pbqg = (const float*)d_in[20];
  const float* pbv = (const float*)d_in[21];
  const float* pbvg = (const float*)d_in[22];
  const float* pln1b = (const float*)d_in[23];
  const float* pln1g = (const float*)d_in[24];
  const float* pln2b = (const float*)d_in[25];
  const float* pln2g = (const float*)d_in[26];

  const size_t BS = (size_t)Bz * Sq;
  char* wsp = (char*)d_ws;
  size_t off = 0;
  auto carve = [&](size_t bytes) -> char* {
    char* p = wsp + off;
    off += (bytes + 255) & ~(size_t)255;
    return p;
  };
  float* x = (float*)carve(BS * Hd * sizeof(float));          // residual stream
  _Float16* xh = (_Float16*)carve(BS * Hd * 2);               // LN'd input (f16)
  _Float16* qh = (_Float16*)carve(BS * Hd * 2);
  _Float16* kh = (_Float16*)carve(BS * Hd * 2);
  _Float16* vh = (_Float16*)carve(BS * Hd * 2);
  _Float16* qgh = (_Float16*)carve(BS * Hd * 2);
  _Float16* kgh = (_Float16*)carve(BS * Hd * 2);
  _Float16* vgh = (_Float16*)carve(BS * Hd * 2);
  _Float16* ath = (_Float16*)carve(BS * Hd * 2);              // attention out
  _Float16* hmid = (_Float16*)carve(BS * Ff * 2);             // FFN mid (f16)
  float* am = (float*)carve(BS * sizeof(float));

  auto gemm = [&](int mode, const _Float16* A, const float* Wt,
                  const float* bias, float* oF, _Float16* oH, const float* res,
                  int M, int N, int K, float sc) {
    dim3 g(N / 64, M / 64);
    switch (mode) {
      case 0: gemm_kernel<0><<<g, 256, 0, stream>>>(A, Wt, bias, oF, oH, res, M, N, K, sc); break;
      case 1: gemm_kernel<1><<<g, 256, 0, stream>>>(A, Wt, bias, oF, oH, res, M, N, K, sc); break;
      case 2: gemm_kernel<2><<<g, 256, 0, stream>>>(A, Wt, bias, oF, oH, res, M, N, K, sc); break;
      default: gemm_kernel<3><<<g, 256, 0, stream>>>(A, Wt, bias, oF, oH, res, M, N, K, sc); break;
    }
  };

  const float qscale = 0.125f;  // 1/sqrt(D)

  am_kernel<<<(int)((BS + 255) / 256), 256, 0, stream>>>(extm, am, (int)BS);
  embed_kernel<<<(int)BS, 256, 0, stream>>>(sent, mask, x, xh);

  const size_t HH = (size_t)Hd * Hd;
  const size_t HF = (size_t)Hd * Ff;
  for (int l = 0; l < Ll; ++l) {
    if (l != 0)
      ln_kernel<<<(int)BS, 256, 0, stream>>>(x, pln1g + (size_t)l * Hd,
                                             pln1b + (size_t)l * Hd, nullptr,
                                             xh);
    // projections (xh holds xin in f16)
    gemm(1, xh, pWq + l * HH, pbq + (size_t)l * Hd, nullptr, qh, nullptr,
         (int)BS, Hd, Hd, qscale);
    gemm(0, xh, pWk + l * HH, pbk + (size_t)l * Hd, nullptr, kh, nullptr,
         (int)BS, Hd, Hd, 1.0f);
    gemm(0, xh, pWv + l * HH, pbv + (size_t)l * Hd, nullptr, vh, nullptr,
         (int)BS, Hd, Hd, 1.0f);
    gemm(1, xh, pWqg + l * HH, pbqg + (size_t)l * Hd, nullptr, qgh, nullptr,
         (int)BS, Hd, Hd, qscale);
    gemm(0, xh, pWkg + l * HH, pbkg + (size_t)l * Hd, nullptr, kgh, nullptr,
         (int)BS, Hd, Hd, 1.0f);
    gemm(0, xh, pWvg + l * HH, pbvg + (size_t)l * Hd, nullptr, vgh, nullptr,
         (int)BS, Hd, Hd, 1.0f);

    local_attn_kernel<<<dim3(nCh, NHh, Bz), 256, 0, stream>>>(qh, kh, vh, am,
                                                              ath);
    global_attn_kernel<<<dim3(NHh, Bz), 128, 0, stream>>>(qgh, kgh, vgh, am,
                                                          ath);
    // output projection + residual (in-place on x)
    gemm(3, ath, pWo + l * HH, pbo + (size_t)l * Hd, x, nullptr, x, (int)BS,
         Hd, Hd, 1.0f);
    // FFN
    ln_kernel<<<(int)BS, 256, 0, stream>>>(x, pln2g + (size_t)l * Hd,
                                           pln2b + (size_t)l * Hd, nullptr, xh);
    gemm(2, xh, pW1 + l * HF, pb1 + (size_t)l * Ff, nullptr, hmid, nullptr,
         (int)BS, Ff, Hd, 1.0f);
    gemm(3, hmid, pW2 + l * HF, pb2 + (size_t)l * Hd, x, nullptr, x, (int)BS,
         Hd, Ff, 1.0f);
  }
  ln_kernel<<<(int)BS, 256, 0, stream>>>(x, final_g, final_b, (float*)d_out,
                                         nullptr);
}
